// H2GCN_EGO_65472481460997
// MI455X (gfx1250) — compile-verified, matrix operand order checked by reading, not run on hardware
//
#include <hip/hip_runtime.h>
#include <hip/hip_bf16.h>

typedef _Float16 v16h __attribute__((ext_vector_type(16)));
typedef float    v8f  __attribute__((ext_vector_type(8)));

// ---------------------------------------------------------------------------
// Utility kernels
// ---------------------------------------------------------------------------
__global__ void zero_kernel(float4* __restrict__ p, long long n4) {
  long long i = blockIdx.x * (long long)blockDim.x + threadIdx.x;
  if (i < n4) p[i] = make_float4(0.f, 0.f, 0.f, 0.f);
}

__global__ void deg_kernel(const int* __restrict__ ei, long long E,
                           float* __restrict__ deg) {
  long long e = blockIdx.x * (long long)blockDim.x + threadIdx.x;
  if (e < E) atomicAdd(&deg[ei[E + e]], 1.0f);
}

__global__ void dinv_kernel(float* __restrict__ d, long long n) {
  long long i = blockIdx.x * (long long)blockDim.x + threadIdx.x;
  if (i < n) d[i] = rsqrtf(d[i] + 1.0f);   // self-loop: deg+1
}

// ---------------------------------------------------------------------------
// Weight pre-pack: W[K,N] (f32, row-major) -> f16 WMMA B-fragments.
// Fragment t = ((kt*NT)+nt)*32 + lane holds 16 contiguous halves matching the
// CDNA5 16-bit B layout: lane&15 = N-in-tile, lane>>4 selects K-octet half.
// ---------------------------------------------------------------------------
__global__ void pack_b_kernel(const float* __restrict__ B, int N, int KT, int NT,
                              _Float16* __restrict__ out) {
  int t = blockIdx.x * blockDim.x + threadIdx.x;
  int total = KT * NT * 32;
  if (t >= total) return;
  int lane = t & 31;
  int nt   = (t >> 5) % NT;
  int kt   = t / (32 * NT);
  int n    = nt * 16 + (lane & 15);
  int half = lane >> 4;
  _Float16* o = out + (size_t)t * 16;
#pragma unroll
  for (int j = 0; j < 8; ++j)
    o[j]     = (_Float16)B[(kt * 32 + half * 8 + j) * N + n];
#pragma unroll
  for (int j = 0; j < 8; ++j)
    o[8 + j] = (_Float16)B[(kt * 32 + 16 + half * 8 + j) * N + n];
}

// ---------------------------------------------------------------------------
// Async-stage packed B fragments into LDS (ASYNCcnt path).
// Low 32 bits of a generic shared-memory address are the LDS byte offset.
// ---------------------------------------------------------------------------
template <int BYTES>
__device__ __forceinline__ void stage_b_async(const _Float16* __restrict__ Bp,
                                              _Float16* smem, int lane) {
  unsigned lbase = (unsigned)(unsigned long long)(uintptr_t)&smem[0];
  const char* g = (const char*)Bp;
#pragma unroll
  for (int ofs = 0; ofs < BYTES; ofs += 512) {       // 32 lanes x 16B per step
    unsigned loff = lbase + (unsigned)(ofs + lane * 16);
    unsigned long long ga = (unsigned long long)(uintptr_t)(g + ofs + lane * 16);
    asm volatile("global_load_async_to_lds_b128 %0, %1, off"
                 :: "v"(loff), "v"(ga) : "memory");
  }
  asm volatile("s_wait_asynccnt 0x0" ::: "memory");
}

// ---------------------------------------------------------------------------
// WMMA A-tile loader (f32 memory -> f16 fragment, CDNA5 wave32 layout).
// ---------------------------------------------------------------------------
__device__ __forceinline__ v16h load_a_tile(const float* __restrict__ A, int lda,
                                            int mBase, int k0, int lane) {
  int m    = mBase + (lane & 15);
  int half = lane >> 4;
  const float* row = A + (long long)m * lda;
  int kA = k0 + half * 8;
  int kB = k0 + 16 + half * 8;
  float4 u0 = *(const float4*)(row + kA);
  float4 u1 = *(const float4*)(row + kA + 4);
  float4 u2 = *(const float4*)(row + kB);
  float4 u3 = *(const float4*)(row + kB + 4);
  v16h a;
  a[0]  = (_Float16)u0.x; a[1]  = (_Float16)u0.y; a[2]  = (_Float16)u0.z; a[3]  = (_Float16)u0.w;
  a[4]  = (_Float16)u1.x; a[5]  = (_Float16)u1.y; a[6]  = (_Float16)u1.z; a[7]  = (_Float16)u1.w;
  a[8]  = (_Float16)u2.x; a[9]  = (_Float16)u2.y; a[10] = (_Float16)u2.z; a[11] = (_Float16)u2.w;
  a[12] = (_Float16)u3.x; a[13] = (_Float16)u3.y; a[14] = (_Float16)u3.z; a[15] = (_Float16)u3.w;
  return a;
}

// ---------------------------------------------------------------------------
// GEMM: one wave per 16-row strip; all NT n-tiles accumulated per A fragment.
// C[M, 16*NT] = A[M, 32*KTILES] @ W, with W pre-packed as f16 fragments.
// ---------------------------------------------------------------------------
template <int KTILES, int NT>
__global__ __launch_bounds__(32)
void gemm_wmma_packed_kernel(const float* __restrict__ A,
                             const _Float16* __restrict__ Bp,
                             float* __restrict__ C, int lda) {
  __shared__ _Float16 smem[KTILES * NT * 32 * 16];
  int lane = threadIdx.x & 31;
  stage_b_async<KTILES * NT * 32 * 16 * 2>(Bp, smem, lane);

  int mBase = blockIdx.x * 16;
  v8f z = {};
  v8f acc[NT];
#pragma unroll
  for (int nt = 0; nt < NT; ++nt) acc[nt] = z;

#pragma unroll
  for (int kt = 0; kt < KTILES; ++kt) {
    v16h a = load_a_tile(A, lda, mBase, kt * 32, lane);
#pragma unroll
    for (int nt = 0; nt < NT; ++nt) {
      v16h b = *(const v16h*)&smem[(((kt * NT) + nt) * 32 + lane) * 16];
      acc[nt] = __builtin_amdgcn_wmma_f32_16x16x32_f16(false, a, false, b,
                                                       (short)0, acc[nt], false, false);
    }
  }

  const int Ncols = 16 * NT;
  int half = lane >> 4;
#pragma unroll
  for (int nt = 0; nt < NT; ++nt) {
    int col = nt * 16 + (lane & 15);
#pragma unroll
    for (int r = 0; r < 8; ++r) {
      int row = mBase + half * 8 + r;
      C[(long long)row * Ncols + col] = acc[nt][r];
    }
  }
}

// Final linear: logits = concat(x[128], h1[64], h2[32]) @ Wl[224,32] + bl.
// KTILES=7, NT=2; K-step boundaries (128, 192) align with 32-wide steps.
__global__ __launch_bounds__(32)
void gemm_cat_wmma_kernel(const float* __restrict__ x,
                          const float* __restrict__ h1,
                          const float* __restrict__ h2,
                          const _Float16* __restrict__ Bp,
                          const float* __restrict__ bl,
                          float* __restrict__ out) {
  __shared__ _Float16 smem[7 * 2 * 32 * 16];
  int lane = threadIdx.x & 31;
  stage_b_async<7 * 2 * 32 * 16 * 2>(Bp, smem, lane);

  int mBase = blockIdx.x * 16;
  v8f acc0 = {}, acc1 = {};
#pragma unroll
  for (int kt = 0; kt < 7; ++kt) {
    int k0 = kt * 32;
    const float* src; int ld; int kl;
    if (k0 < 128)      { src = x;  ld = 128; kl = k0; }
    else if (k0 < 192) { src = h1; ld = 64;  kl = k0 - 128; }
    else               { src = h2; ld = 32;  kl = k0 - 192; }
    v16h a  = load_a_tile(src, ld, mBase, kl, lane);
    v16h b0 = *(const v16h*)&smem[((kt * 2 + 0) * 32 + lane) * 16];
    acc0 = __builtin_amdgcn_wmma_f32_16x16x32_f16(false, a, false, b0,
                                                  (short)0, acc0, false, false);
    v16h b1 = *(const v16h*)&smem[((kt * 2 + 1) * 32 + lane) * 16];
    acc1 = __builtin_amdgcn_wmma_f32_16x16x32_f16(false, a, false, b1,
                                                  (short)0, acc1, false, false);
  }

  int half = lane >> 4;
  int c0 = (lane & 15), c1 = 16 + (lane & 15);
  float bias0 = bl[c0], bias1 = bl[c1];
#pragma unroll
  for (int r = 0; r < 8; ++r) {
    int row = mBase + half * 8 + r;
    out[(long long)row * 32 + c0] = acc0[r] + bias0;
    out[(long long)row * 32 + c1] = acc1[r] + bias1;
  }
}

// ---------------------------------------------------------------------------
// Edge scatter: agg[dst] += h[src] * dinv[src]*dinv[dst].
// (1 << cshift) lanes per edge; each lane handles one contiguous float4.
// ---------------------------------------------------------------------------
__global__ void scatter_kernel(const float* __restrict__ h,
                               const float* __restrict__ dinv,
                               const int* __restrict__ ei,
                               long long E, int F, int cshift,
                               float* __restrict__ agg) {
  long long tid   = blockIdx.x * (long long)blockDim.x + threadIdx.x;
  long long total = E << cshift;
  if (tid >= total) return;
  long long e = tid >> cshift;
  int c = (int)(tid & ((1 << cshift) - 1));
  int s = ei[e];
  int d = ei[E + e];
  __builtin_prefetch(&ei[e + 4096], 0, 1);           // global_prefetch_b8
  __builtin_prefetch(&ei[E + e + 4096], 0, 1);
  float nrm = dinv[s] * dinv[d];
  float4 v = ((const float4*)(h + (long long)s * F))[c];
  float* outp = agg + (long long)d * F + 4 * c;
  atomicAdd(outp + 0, v.x * nrm);
  atomicAdd(outp + 1, v.y * nrm);
  atomicAdd(outp + 2, v.z * nrm);
  atomicAdd(outp + 3, v.w * nrm);
}

// agg = (agg + hpre*dinv^2 + bias)  [+ ReLU]
__global__ void finalize_kernel(float* __restrict__ agg,
                                const float* __restrict__ hpre,
                                const float* __restrict__ dinv,
                                const float* __restrict__ bias,
                                long long total, int fshift, int fmask, int relu) {
  long long i = blockIdx.x * (long long)blockDim.x + threadIdx.x;
  if (i >= total) return;
  long long row = i >> fshift;
  int col = (int)(i & fmask);
  float di = dinv[row];
  float v = agg[i] + hpre[i] * di * di + bias[col];
  agg[i] = relu ? fmaxf(v, 0.0f) : v;
}

// In-place log_softmax over 32 columns; one wave32 per row, one lane per class.
__global__ void log_softmax_kernel(float* __restrict__ out, int nrows) {
  int lane = threadIdx.x & 31;
  int row  = blockIdx.x * (blockDim.x >> 5) + (threadIdx.x >> 5);
  if (row >= nrows) return;
  float v = out[(long long)row * 32 + lane];
  float m = v;
#pragma unroll
  for (int o = 16; o > 0; o >>= 1) m = fmaxf(m, __shfl_xor(m, o, 32));
  float e = __expf(v - m);
  float s = e;
#pragma unroll
  for (int o = 16; o > 0; o >>= 1) s += __shfl_xor(s, o, 32);
  out[(long long)row * 32 + lane] = (v - m) - __logf(s);
}

// ---------------------------------------------------------------------------
// Launcher
// ---------------------------------------------------------------------------
extern "C" void kernel_launch(void* const* d_in, const int* in_sizes, int n_in,
                              void* d_out, int out_size, void* d_ws, size_t ws_size,
                              hipStream_t stream) {
  const float* x  = (const float*)d_in[0];
  const int*   ei = (const int*)d_in[1];
  const float* W1 = (const float*)d_in[2];
  const float* b1 = (const float*)d_in[3];
  const float* W2 = (const float*)d_in[4];
  const float* b2 = (const float*)d_in[5];
  const float* Wl = (const float*)d_in[6];
  const float* bl = (const float*)d_in[7];
  float* out = (float*)d_out;

  const int IN_C = 128, HID_C = 64, OUT_C = 32;
  const long long nNodes = in_sizes[0] / IN_C;   // 100000
  const long long E      = in_sizes[1] / 2;      // 1.6M

  // Workspace layout (floats then packed f16), all regions >=16B aligned.
  float* ws    = (float*)d_ws;
  size_t nPad  = (size_t)((nNodes + 255) & ~255LL);
  float* dinv  = ws;                              // nNodes (deg -> dinv in place)
  float* hpre1 = ws + nPad;                       // nNodes * 64
  float* h1    = hpre1 + (size_t)nNodes * HID_C;  // nNodes * 64 (agg -> h1)
  float* hpre2 = h1    + (size_t)nNodes * HID_C;  // nNodes * 32
  float* h2    = hpre2 + (size_t)nNodes * OUT_C;  // nNodes * 32 (agg -> h2)
  _Float16* Bp1 = (_Float16*)(h2 + (size_t)nNodes * OUT_C);  // 4*4*512 halves
  _Float16* Bp2 = Bp1 + 4 * 4 * 512;                         // 2*2*512 halves
  _Float16* Bpl = Bp2 + 2 * 2 * 512;                         // 7*2*512 halves

  auto cdiv = [](long long a, long long b) { return (unsigned)((a + b - 1) / b); };

  // 1) zero accumulators
  zero_kernel<<<cdiv(nNodes / 4, 256), 256, 0, stream>>>((float4*)dinv, nNodes / 4);
  zero_kernel<<<cdiv(nNodes * HID_C / 4, 256), 256, 0, stream>>>((float4*)h1, nNodes * HID_C / 4);
  zero_kernel<<<cdiv(nNodes * OUT_C / 4, 256), 256, 0, stream>>>((float4*)h2, nNodes * OUT_C / 4);

  // 2) degrees + dinv ; pre-pack weights into f16 WMMA fragments
  deg_kernel<<<cdiv(E, 256), 256, 0, stream>>>(ei, E, dinv);
  dinv_kernel<<<cdiv(nNodes, 256), 256, 0, stream>>>(dinv, nNodes);
  pack_b_kernel<<<cdiv(4 * 4 * 32, 256), 256, 0, stream>>>(W1, HID_C, 4, 4, Bp1);
  pack_b_kernel<<<cdiv(2 * 2 * 32, 256), 256, 0, stream>>>(W2, OUT_C, 2, 2, Bp2);
  pack_b_kernel<<<cdiv(7 * 2 * 32, 256), 256, 0, stream>>>(Wl, OUT_C, 7, 2, Bpl);

  // 3) layer 1: hpre1 = x @ W1 ; scatter ; h1 = relu(agg + hpre1*dinv^2 + b1)
  gemm_wmma_packed_kernel<4, 4><<<(unsigned)(nNodes / 16), 32, 0, stream>>>(
      x, Bp1, hpre1, IN_C);
  scatter_kernel<<<cdiv(E * 16, 256), 256, 0, stream>>>(hpre1, dinv, ei, E, HID_C, 4, h1);
  finalize_kernel<<<cdiv(nNodes * HID_C, 256), 256, 0, stream>>>(
      h1, hpre1, dinv, b1, nNodes * HID_C, 6, 63, 1);

  // 4) layer 2: hpre2 = h1 @ W2 ; scatter ; h2 = agg + hpre2*dinv^2 + b2
  gemm_wmma_packed_kernel<2, 2><<<(unsigned)(nNodes / 16), 32, 0, stream>>>(
      h1, Bp2, hpre2, HID_C);
  scatter_kernel<<<cdiv(E * 8, 256), 256, 0, stream>>>(hpre2, dinv, ei, E, OUT_C, 3, h2);
  finalize_kernel<<<cdiv(nNodes * OUT_C, 256), 256, 0, stream>>>(
      h2, hpre2, dinv, b2, nNodes * OUT_C, 5, 31, 0);

  // 5) logits = [x|h1|h2] @ Wl + bl ; in-place log_softmax
  gemm_cat_wmma_kernel<<<(unsigned)(nNodes / 16), 32, 0, stream>>>(
      x, h1, h2, Bpl, bl, out);
  log_softmax_kernel<<<cdiv(nNodes * 32, 256), 256, 0, stream>>>(out, (int)nNodes);
}
